// multihead_attation_block_5042291606103
// MI455X (gfx1250) — compile-verified, matrix operand order checked by reading, not run on hardware
//
#include <hip/hip_runtime.h>
#include <hip/hip_bf16.h>

// ---------------------------------------------------------------------------
// MHA block on gfx1250: all GEMMs via v_wmma_f32_16x16x32_bf16, all LDS tile
// fills via the Tensor Data Mover (tensor_load_to_lds), double-buffered so
// the TDM overlaps WMMA compute.  B=4, S=2048, D=1024, H=16, hd=64.
// ---------------------------------------------------------------------------

#define BATCH  4
#define SEQ    2048
#define DMODEL 1024
#define NHEAD  16
#define HD     64
#define ROWS   (BATCH * SEQ)        // 8192

typedef __attribute__((ext_vector_type(16))) __bf16 v16bf;
typedef __attribute__((ext_vector_type(8)))  __bf16 v8bf;
typedef __attribute__((ext_vector_type(8)))  float  v8f;
typedef __attribute__((ext_vector_type(4)))  float  v4f;

#if __has_builtin(__builtin_amdgcn_tensor_load_to_lds)
#define HAVE_TDM 1
typedef __attribute__((ext_vector_type(4))) unsigned int v4u;
typedef __attribute__((ext_vector_type(8))) int          v8i;
typedef __attribute__((ext_vector_type(4))) int          v4i;
#elif __has_builtin(__builtin_amdgcn_global_load_async_to_lds_b128)
#define HAVE_ASYNC_LDS 1
#endif

static __device__ inline v8f wmma_bf16(v16bf a, v16bf b, v8f c) {
    return __builtin_amdgcn_wmma_f32_16x16x32_bf16(false, a, false, b,
                                                   (short)0, c, false, false);
}

// 16x32 A/B fragment load from row-major tile, row stride `ld` elements.
// lane L: row (L&15); elems 0..7 = k0+8*(L>>4)+j, elems 8..15 = +16.
static __device__ inline v16bf frag_ld(const __bf16* base, int row0, int ld, int k0) {
    int lane = threadIdx.x & 31;
    const __bf16* p = base + (size_t)(row0 + (lane & 15)) * ld + k0 + ((lane >> 4) << 3);
    v8bf lo = *(const v8bf*)(p);
    v8bf hi = *(const v8bf*)(p + 16);
    v16bf f;
#pragma unroll
    for (int i = 0; i < 8; ++i) { f[i] = lo[i]; f[i + 8] = hi[i]; }
    return f;
}

static __device__ inline float rmax16(float v) {
#pragma unroll
    for (int m = 1; m < 16; m <<= 1) v = fmaxf(v, __shfl_xor(v, m, 32));
    return v;
}
static __device__ inline float rsum16(float v) {
#pragma unroll
    for (int m = 1; m < 16; m <<= 1) v += __shfl_xor(v, m, 32);
    return v;
}

// ---------------------------------------------------------------------------
// TDM: load a 2-D bf16 tile (rows x 64 elems) into LDS with +8-elem row pad
// (pad 4 DWORDs every 32 data DWORDs -> LDS pitch 144 B = 72 bf16).
// D# layout per CDNA5 ISA sec 8.3-8.6.  Issued by one wave; EXEC ignored.
// ---------------------------------------------------------------------------
#if defined(HAVE_TDM)
static __device__ inline void tdm_load64(const __bf16* gsrc, unsigned lds_byte_off,
                                         int rows, int row_stride_elems) {
    unsigned long long ga = (unsigned long long)(uintptr_t)gsrc;
    v4u g0;
    g0[0] = 1u;                                            // count=1 (valid user D#)
    g0[1] = lds_byte_off;                                  // lds_addr
    g0[2] = (unsigned)ga;                                  // global_addr[31:0]
    g0[3] = (unsigned)((ga >> 32) & 0x01FFFFFFull)         // global_addr[56:32]
          | (2u << 30);                                    // type=2 (image)
    v8i g1;
    g1[0] = (int)((1u << 16)                               // data_size = 2 bytes
                | (1u << 20)                               // pad_enable
                | (4u << 22)                               // pad_interval: 32 DWORDs
                | (3u << 25));                             // pad_amount: 4 DWORDs
    g1[1] = (int)(64u << 16);                              // tensor_dim0 = 64 (lo16)
    g1[2] = (int)(((unsigned)rows & 0xFFFFu) << 16);       // tensor_dim1 = rows (lo16)
    g1[3] = (int)(64u << 16);                              // tile_dim0 = 64
    g1[4] = (int)((unsigned)rows & 0xFFFFu);               // tile_dim1 = rows, tile_dim2=0
    g1[5] = row_stride_elems;                              // tensor_dim0_stride[31:0]
    g1[6] = 0;
    g1[7] = 0;
    v4i z4 = {0, 0, 0, 0};
#if defined(__clang_major__) && (__clang_major__ >= 23)
    v8i z8 = {0, 0, 0, 0, 0, 0, 0, 0};
    __builtin_amdgcn_tensor_load_to_lds(g0, g1, z4, z4, z8, 0);
#else
    __builtin_amdgcn_tensor_load_to_lds(g0, g1, z4, z4, 0);
#endif
}
static __device__ inline void tdm_wait0() {
#if __has_builtin(__builtin_amdgcn_s_wait_tensorcnt)
    __builtin_amdgcn_s_wait_tensorcnt(0);
#else
    asm volatile("s_wait_tensorcnt 0" ::: "memory");
#endif
}
static __device__ inline void tdm_wait2() {
#if __has_builtin(__builtin_amdgcn_s_wait_tensorcnt)
    __builtin_amdgcn_s_wait_tensorcnt(2);
#else
    asm volatile("s_wait_tensorcnt 2" ::: "memory");
#endif
}
#else
// fallback 16-byte global->LDS copy (async if available, else sync vector)
static __device__ inline void cp16_g2l(__bf16* ldst, const __bf16* gsrc) {
#if defined(HAVE_ASYNC_LDS)
    __builtin_amdgcn_global_load_async_to_lds_b128(
        (const __attribute__((address_space(1))) void*)gsrc,
        (__attribute__((address_space(3))) void*)ldst, 0, 0);
#else
    *(v8bf*)ldst = *(const v8bf*)gsrc;
#endif
}
static __device__ inline void cp_fence() {
#if defined(HAVE_ASYNC_LDS)
#if __has_builtin(__builtin_amdgcn_s_wait_asynccnt)
    __builtin_amdgcn_s_wait_asynccnt(0);
#else
    asm volatile("s_wait_asynccnt 0" ::: "memory");
#endif
#endif
}
#endif

// ---------------------------------------------------------------------------
// Prep kernels: q -> bf16; W -> bf16 transposed [n][k].
// ---------------------------------------------------------------------------
__global__ __launch_bounds__(256) void cvt_q_kernel(const float* __restrict__ X,
                                                    __bf16* __restrict__ Y) {
    size_t i = ((size_t)blockIdx.x * 256 + threadIdx.x) * 8;
    v4f a = *(const v4f*)(X + i);
    v4f b = *(const v4f*)(X + i + 4);
    v8bf o;
#pragma unroll
    for (int j = 0; j < 4; ++j) { o[j] = (__bf16)a[j]; o[4 + j] = (__bf16)b[j]; }
    *(v8bf*)(Y + i) = o;
}

__global__ __launch_bounds__(256) void cvt_w_kernel(const float* __restrict__ W,
                                                    __bf16* __restrict__ WT) {
    __shared__ float t[32][33];
    const int k0 = blockIdx.x * 32, n0 = blockIdx.y * 32;
    const int tx = threadIdx.x & 31, ty = threadIdx.x >> 5;   // 8 rows/pass
#pragma unroll
    for (int i = 0; i < 4; ++i)
        t[ty + 8 * i][tx] = W[(size_t)(k0 + ty + 8 * i) * DMODEL + n0 + tx];
    __syncthreads();
#pragma unroll
    for (int i = 0; i < 4; ++i)
        WT[(size_t)(n0 + ty + 8 * i) * DMODEL + k0 + tx] = (__bf16)t[tx][ty + 8 * i];
}

// ---------------------------------------------------------------------------
// Kernel 1: fused QKV projection.  qb[8192,1024] bf16 @ WT (pre-transposed)
// -> bf16 head-major.  Q scaled by 1/8; V stored transposed [b,h][d][s].
// Block tile 128x128, K-step 64, 8 waves (2x4), double-buffered TDM.
// ---------------------------------------------------------------------------
__global__ __launch_bounds__(256) void proj_qkv_kernel(
    const __bf16* __restrict__ qb,
    const __bf16* __restrict__ WTq, const float* __restrict__ bq,
    const __bf16* __restrict__ WTk, const float* __restrict__ bk,
    const __bf16* __restrict__ WTv, const float* __restrict__ bv,
    __bf16* __restrict__ Qp, __bf16* __restrict__ Kp, __bf16* __restrict__ Vp)
{
    const __bf16* WT; const float* bias; __bf16* Out; float scale = 1.0f;
    const int z = blockIdx.z;
    if (z == 0)      { WT = WTq; bias = bq; Out = Qp; scale = 0.125f; }
    else if (z == 1) { WT = WTk; bias = bk; Out = Kp; }
    else             { WT = WTv; bias = bv; Out = Vp; }

    __shared__ __bf16 sA[2][128 * 72];   // [m][k] ld=72, double buffered
    __shared__ __bf16 sB[2][128 * 72];   // [n][k] ld=72

    const int tid  = threadIdx.x;
    const int lane = tid & 31;
    const int w    = tid >> 5;
    const int wm   = w & 1;
    const int wn   = w >> 1;
    const int m0   = blockIdx.x * 128;
    const int n0   = blockIdx.y * 128;
    const int NT   = DMODEL / 64;

    v8f acc[4][2];
#pragma unroll
    for (int ms = 0; ms < 4; ++ms)
#pragma unroll
        for (int ns = 0; ns < 2; ++ns)
#pragma unroll
            for (int r = 0; r < 8; ++r) acc[ms][ns][r] = 0.0f;

#if defined(HAVE_TDM)
    if (w == 0) {
        tdm_load64(qb + (size_t)m0 * DMODEL, (unsigned)(uintptr_t)sA[0], 128, DMODEL);
        tdm_load64(WT + (size_t)n0 * DMODEL, (unsigned)(uintptr_t)sB[0], 128, DMODEL);
    }
#endif

    for (int kt = 0; kt < NT; ++kt) {
        const int buf = kt & 1;
#if defined(HAVE_TDM)
        if (w == 0) {
            if (kt + 1 < NT) {
                tdm_load64(qb + (size_t)m0 * DMODEL + (kt + 1) * 64,
                           (unsigned)(uintptr_t)sA[buf ^ 1], 128, DMODEL);
                tdm_load64(WT + (size_t)n0 * DMODEL + (kt + 1) * 64,
                           (unsigned)(uintptr_t)sB[buf ^ 1], 128, DMODEL);
                tdm_wait2();          // current pair done, next pair in flight
            } else {
                tdm_wait0();
            }
        }
#else
#pragma unroll
        for (int i = 0; i < 4; ++i) {
            int idx = tid + 256 * i;                 // 1024 16B chunks
            int r = idx >> 3, c = (idx & 7) << 3;
            cp16_g2l(&sA[buf][r * 72 + c], qb + (size_t)(m0 + r) * DMODEL + kt * 64 + c);
            cp16_g2l(&sB[buf][r * 72 + c], WT + (size_t)(n0 + r) * DMODEL + kt * 64 + c);
        }
        cp_fence();
#endif
        __syncthreads();             // publish buf to all waves

        const __bf16* cA = sA[buf];
        const __bf16* cB = sB[buf];
#pragma unroll
        for (int k0 = 0; k0 < 64; k0 += 32) {
            v16bf bf[2];
#pragma unroll
            for (int ns = 0; ns < 2; ++ns)
                bf[ns] = frag_ld(cB, wn * 32 + ns * 16, 72, k0);
#pragma unroll
            for (int ms = 0; ms < 4; ++ms) {
                v16bf af = frag_ld(cA, wm * 64 + ms * 16, 72, k0);
#pragma unroll
                for (int ns = 0; ns < 2; ++ns)
                    acc[ms][ns] = wmma_bf16(af, bf[ns], acc[ms][ns]);
            }
        }
        __syncthreads();             // buf free before it is re-targeted
    }

    const int half = lane >> 4, ln = lane & 15;
#pragma unroll
    for (int ms = 0; ms < 4; ++ms)
#pragma unroll
        for (int ns = 0; ns < 2; ++ns) {
            int ncol = n0 + wn * 32 + ns * 16 + ln;
            int h = ncol >> 6, d = ncol & 63;
            float bv_ = bias[ncol];
#pragma unroll
            for (int r = 0; r < 8; ++r) {
                int m = m0 + wm * 64 + ms * 16 + r + 8 * half;
                int b = m >> 11, s = m & (SEQ - 1);
                float val = (acc[ms][ns][r] + bv_) * scale;
                if (z == 2)   // V transposed: [b,h][d][s]
                    Out[((size_t)(b * NHEAD + h) * HD + d) * SEQ + s] = (__bf16)val;
                else          // Q,K: [b,h][s][d]
                    Out[((size_t)(b * NHEAD + h) * SEQ + s) * HD + d] = (__bf16)val;
            }
        }
}

// ---------------------------------------------------------------------------
// Kernel 2: flash attention per (b,h).  128 queries/block, 8 waves x 16 rows,
// key blocks of 64.  K tile [key][d], V tile [d][key] (pre-transposed in
// memory), double-buffered TDM; P staged via wave-private LDS.
// ---------------------------------------------------------------------------
__global__ __launch_bounds__(256) void attn_kernel(
    const __bf16* __restrict__ Qp, const __bf16* __restrict__ Kp,
    const __bf16* __restrict__ Vp, __bf16* __restrict__ Xa)
{
    const int bh   = blockIdx.y;
    const int b    = bh >> 4;
    const int h    = bh & (NHEAD - 1);
    const int tid  = threadIdx.x;
    const int lane = tid & 31;
    const int w    = tid >> 5;
    const int qbase = blockIdx.x * 128 + w * 16;
    const int NB   = SEQ / 64;

    const __bf16* Qh = Qp + (size_t)bh * SEQ * HD;
    const __bf16* Kh = Kp + (size_t)bh * SEQ * HD;
    const __bf16* Vh = Vp + (size_t)bh * HD * SEQ;   // [d][s]

    __shared__ __bf16 sK[2][64 * 72];       // [key][d]
    __shared__ __bf16 sV[2][64 * 72];       // [d][key]
    __shared__ __bf16 sP[8][16 * 72];       // per-wave P [m][key]

    v16bf qf0 = frag_ld(Qh, qbase, HD, 0);
    v16bf qf1 = frag_ld(Qh, qbase, HD, 32);

    v8f o[4];
    float M[8], L[8];
#pragma unroll
    for (int nt = 0; nt < 4; ++nt)
#pragma unroll
        for (int r = 0; r < 8; ++r) o[nt][r] = 0.0f;
#pragma unroll
    for (int r = 0; r < 8; ++r) { M[r] = -3.0e38f; L[r] = 0.0f; }

    const int half = lane >> 4, ln = lane & 15;

#if defined(HAVE_TDM)
    if (w == 0) {
        tdm_load64(Kh, (unsigned)(uintptr_t)sK[0], 64, HD);
        tdm_load64(Vh, (unsigned)(uintptr_t)sV[0], 64, SEQ);
    }
#endif

    for (int kb = 0; kb < NB; ++kb) {
        const int key0 = kb * 64;
        const int buf  = kb & 1;
#if defined(HAVE_TDM)
        if (w == 0) {
            if (kb + 1 < NB) {
                tdm_load64(Kh + (size_t)(key0 + 64) * HD, (unsigned)(uintptr_t)sK[buf ^ 1], 64, HD);
                tdm_load64(Vh + key0 + 64,                (unsigned)(uintptr_t)sV[buf ^ 1], 64, SEQ);
                tdm_wait2();
            } else {
                tdm_wait0();
            }
        }
#else
#pragma unroll
        for (int i = 0; i < 2; ++i) {
            int idx = tid + 256 * i;                 // 512 16B chunks each
            int r = idx >> 3, c = (idx & 7) << 3;
            cp16_g2l(&sK[buf][r * 72 + c], Kh + (size_t)(key0 + r) * HD + c);
            cp16_g2l(&sV[buf][r * 72 + c], Vh + (size_t)r * SEQ + key0 + c);
        }
        cp_fence();
#endif
        __syncthreads();

        const __bf16* cK = sK[buf];
        const __bf16* cV = sV[buf];

        // scores = Q K^T : 4 tiles, k = hd = 64
        v8f st[4];
#pragma unroll
        for (int nt = 0; nt < 4; ++nt) {
            v8f s;
#pragma unroll
            for (int r = 0; r < 8; ++r) s[r] = 0.0f;
            s = wmma_bf16(qf0, frag_ld(cK, nt * 16, 72, 0),  s);
            s = wmma_bf16(qf1, frag_ld(cK, nt * 16, 72, 32), s);
            st[nt] = s;
        }

        // online softmax
        float alpha[8];
#pragma unroll
        for (int r = 0; r < 8; ++r) {
            float pm = st[0][r];
#pragma unroll
            for (int nt = 1; nt < 4; ++nt) pm = fmaxf(pm, st[nt][r]);
            pm = rmax16(pm);
            float mnew = fmaxf(M[r], pm);
            alpha[r] = __expf(M[r] - mnew);
            M[r] = mnew;
        }
        float rs[8];
#pragma unroll
        for (int r = 0; r < 8; ++r) rs[r] = 0.0f;
#pragma unroll
        for (int nt = 0; nt < 4; ++nt)
#pragma unroll
            for (int r = 0; r < 8; ++r) {
                float p = __expf(st[nt][r] - M[r]);
                st[nt][r] = p;
                rs[r] += p;
            }
#pragma unroll
        for (int r = 0; r < 8; ++r) L[r] = L[r] * alpha[r] + rsum16(rs[r]);

        __bf16* pw = &sP[w][0];
#pragma unroll
        for (int nt = 0; nt < 4; ++nt)
#pragma unroll
            for (int r = 0; r < 8; ++r)
                pw[(r + 8 * half) * 72 + nt * 16 + ln] = (__bf16)st[nt][r];
#pragma unroll
        for (int nt = 0; nt < 4; ++nt)
#pragma unroll
            for (int r = 0; r < 8; ++r) o[nt][r] *= alpha[r];

        // O += P V
#pragma unroll
        for (int ks = 0; ks < 2; ++ks) {
            v16bf pa = frag_ld(pw, 0, 72, ks * 32);
#pragma unroll
            for (int nt = 0; nt < 4; ++nt)
                o[nt] = wmma_bf16(pa, frag_ld(cV, nt * 16, 72, ks * 32), o[nt]);
        }
        __syncthreads();
    }

    float inv[8];
#pragma unroll
    for (int r = 0; r < 8; ++r) inv[r] = 1.0f / L[r];
#pragma unroll
    for (int nt = 0; nt < 4; ++nt)
#pragma unroll
        for (int r = 0; r < 8; ++r) {
            int m = qbase + r + 8 * half;
            int d = nt * 16 + ln;
            Xa[((size_t)(b * SEQ + m)) * DMODEL + h * HD + d] =
                (__bf16)(o[nt][r] * inv[r]);
        }
}

// ---------------------------------------------------------------------------
// Kernel 3: output projection.  Xa bf16 @ WTo (pre-transposed bf16) + bo
// -> fp32 out.  Double-buffered TDM.
// ---------------------------------------------------------------------------
__global__ __launch_bounds__(256) void out_proj_kernel(
    const __bf16* __restrict__ A,
    const __bf16* __restrict__ WTo, const float* __restrict__ bo,
    float* __restrict__ Out)
{
    __shared__ __bf16 sA[2][128 * 72];
    __shared__ __bf16 sB[2][128 * 72];

    const int tid  = threadIdx.x;
    const int lane = tid & 31;
    const int w    = tid >> 5;
    const int wm   = w & 1;
    const int wn   = w >> 1;
    const int m0   = blockIdx.x * 128;
    const int n0   = blockIdx.y * 128;
    const int NT   = DMODEL / 64;

    v8f acc[4][2];
#pragma unroll
    for (int ms = 0; ms < 4; ++ms)
#pragma unroll
        for (int ns = 0; ns < 2; ++ns)
#pragma unroll
            for (int r = 0; r < 8; ++r) acc[ms][ns][r] = 0.0f;

#if defined(HAVE_TDM)
    if (w == 0) {
        tdm_load64(A   + (size_t)m0 * DMODEL, (unsigned)(uintptr_t)sA[0], 128, DMODEL);
        tdm_load64(WTo + (size_t)n0 * DMODEL, (unsigned)(uintptr_t)sB[0], 128, DMODEL);
    }
#endif

    for (int kt = 0; kt < NT; ++kt) {
        const int buf = kt & 1;
#if defined(HAVE_TDM)
        if (w == 0) {
            if (kt + 1 < NT) {
                tdm_load64(A   + (size_t)m0 * DMODEL + (kt + 1) * 64,
                           (unsigned)(uintptr_t)sA[buf ^ 1], 128, DMODEL);
                tdm_load64(WTo + (size_t)n0 * DMODEL + (kt + 1) * 64,
                           (unsigned)(uintptr_t)sB[buf ^ 1], 128, DMODEL);
                tdm_wait2();
            } else {
                tdm_wait0();
            }
        }
#else
#pragma unroll
        for (int i = 0; i < 4; ++i) {
            int idx = tid + 256 * i;
            int r = idx >> 3, c = (idx & 7) << 3;
            cp16_g2l(&sA[buf][r * 72 + c], A   + (size_t)(m0 + r) * DMODEL + kt * 64 + c);
            cp16_g2l(&sB[buf][r * 72 + c], WTo + (size_t)(n0 + r) * DMODEL + kt * 64 + c);
        }
        cp_fence();
#endif
        __syncthreads();

        const __bf16* cA = sA[buf];
        const __bf16* cB = sB[buf];
#pragma unroll
        for (int k0 = 0; k0 < 64; k0 += 32) {
            v16bf bf[2];
#pragma unroll
            for (int ns = 0; ns < 2; ++ns)
                bf[ns] = frag_ld(cB, wn * 32 + ns * 16, 72, k0);
#pragma unroll
            for (int ms = 0; ms < 4; ++ms) {
                v16bf af = frag_ld(cA, wm * 64 + ms * 16, 72, k0);
#pragma unroll
                for (int ns = 0; ns < 2; ++ns)
                    acc[ms][ns] = wmma_bf16(af, bf[ns], acc[ms][ns]);
            }
        }
        __syncthreads();
    }

    const int half = lane >> 4, ln = lane & 15;
#pragma unroll
    for (int ms = 0; ms < 4; ++ms)
#pragma unroll
        for (int ns = 0; ns < 2; ++ns) {
            int ncol = n0 + wn * 32 + ns * 16 + ln;
            float bv_ = bo[ncol];
#pragma unroll
            for (int r = 0; r < 8; ++r) {
                int m = m0 + wm * 64 + ms * 16 + r + 8 * half;
                Out[(size_t)m * DMODEL + ncol] = acc[ms][ns][r] + bv_;
            }
        }
}

// ---------------------------------------------------------------------------
extern "C" void kernel_launch(void* const* d_in, const int* in_sizes, int n_in,
                              void* d_out, int out_size, void* d_ws, size_t ws_size,
                              hipStream_t stream) {
    const float* q  = (const float*)d_in[0];
    const float* Wq = (const float*)d_in[1];
    const float* bq = (const float*)d_in[2];
    const float* Wk = (const float*)d_in[3];
    const float* bk = (const float*)d_in[4];
    const float* Wv = (const float*)d_in[5];
    const float* bv = (const float*)d_in[6];
    const float* Wo = (const float*)d_in[7];
    const float* bo = (const float*)d_in[8];
    float* out = (float*)d_out;
    (void)in_sizes; (void)n_in; (void)out_size; (void)ws_size;

    const size_t PLANE = (size_t)ROWS * DMODEL;      // 8M bf16
    const size_t WSZ   = (size_t)DMODEL * DMODEL;    // 1M bf16
    __bf16* qb  = (__bf16*)d_ws;
    __bf16* WTq = qb  + PLANE;
    __bf16* WTk = WTq + WSZ;
    __bf16* WTv = WTk + WSZ;
    __bf16* WTo = WTv + WSZ;
    __bf16* Qp  = WTo + WSZ;                         // [b,h][s][d]
    __bf16* Kp  = Qp  + PLANE;                       // [b,h][s][d]
    __bf16* Vp  = Kp  + PLANE;                       // [b,h][d][s]
    __bf16* Xa  = Vp  + PLANE;                       // [b*s][h*d]

    cvt_q_kernel<<<(int)(PLANE / (256 * 8)), 256, 0, stream>>>(q, qb);
    dim3 gt(DMODEL / 32, DMODEL / 32);
    cvt_w_kernel<<<gt, 256, 0, stream>>>(Wq, WTq);
    cvt_w_kernel<<<gt, 256, 0, stream>>>(Wk, WTk);
    cvt_w_kernel<<<gt, 256, 0, stream>>>(Wv, WTv);
    cvt_w_kernel<<<gt, 256, 0, stream>>>(Wo, WTo);

    dim3 g1(ROWS / 128, DMODEL / 128, 3);
    proj_qkv_kernel<<<g1, 256, 0, stream>>>(qb, WTq, bq, WTk, bk, WTv, bv, Qp, Kp, Vp);

    dim3 g2(SEQ / 128, BATCH * NHEAD);
    attn_kernel<<<g2, 256, 0, stream>>>(Qp, Kp, Vp, Xa);

    dim3 g3(ROWS / 128, DMODEL / 128);
    out_proj_kernel<<<g3, 256, 0, stream>>>(Xa, WTo, bo, out);
}
